// WavenetDecoder_24567212933460
// MI455X (gfx1250) — matrix operand order repaired and checked by reading, not compiled, not run on hardware
//
#include <hip/hip_runtime.h>
#include <hip/hip_bf16.h>

// ---------------- types / helpers ----------------
typedef __attribute__((ext_vector_type(16))) __bf16        v16bf;
typedef __attribute__((ext_vector_type(8)))  __bf16        v8bf;
typedef __attribute__((ext_vector_type(8)))  float         v8f;
typedef __attribute__((ext_vector_type(8)))  unsigned short v8us;

#define RW 512
#define SW 256
#define OC 256
#define CD 16
#define TLEN 4096
#define NC 8192          // B*T columns
#define NLAY 30
#define LPAD 520         // LDS column stride in halfs (16B aligned, bank-spread)

__device__ __forceinline__ unsigned short f2bf(float f) {
    unsigned int u = __float_as_uint(f);
    unsigned int r = u + 0x7FFFu + ((u >> 16) & 1u);   // round-to-nearest-even
    return (unsigned short)(r >> 16);
}

union Frag16 { v16bf v; v8bf h[2]; };

// Build a 16-half fragment from two 16-byte chunks (A and B layouts both need
// two contiguous 8-half groups per lane).
__device__ __forceinline__ v16bf ldfrag(const unsigned short* p0, const unsigned short* p1) {
    Frag16 f;
    f.h[0] = *(const v8bf*)p0;
    f.h[1] = *(const v8bf*)p1;
    return f.v;
}

__device__ __forceinline__ v8f zero8() {
    v8f z = {0.f,0.f,0.f,0.f,0.f,0.f,0.f,0.f};
    return z;
}

__device__ __forceinline__ v8f wmma_bf16(v16bf a, v16bf b, v8f c) {
    return __builtin_amdgcn_wmma_f32_16x16x32_bf16(false, a, false, b, (short)0, c, false, false);
}

// ---------------- weight / input conversion kernels ----------------
__global__ void k_cvt_flat(const float* __restrict__ src, unsigned short* __restrict__ dst, int n) {
    for (int i = blockIdx.x * blockDim.x + threadIdx.x; i < n; i += gridDim.x * blockDim.x)
        dst[i] = f2bf(src[i]);
}

// filter/gate: src [L][O][I][K] -> dst [L][K][O][I]  (tap-major so conv = 3 dense K=512 passes)
__global__ void k_cvt_conv(const float* __restrict__ src, unsigned short* __restrict__ dst, int n) {
    for (int idx = blockIdx.x * blockDim.x + threadIdx.x; idx < n; idx += gridDim.x * blockDim.x) {
        int i = idx & 511;
        int r = idx >> 9;
        int o = r & 511; r >>= 9;
        int k = r % 3;
        int l = r / 3;
        dst[idx] = f2bf(src[(((size_t)l * RW + o) * RW + i) * 3 + k]);
    }
}

// cond weights: src [L][O][16] -> dst [L][O][32] (upper 16 zero)
__global__ void k_cvt_cond(const float* __restrict__ src, unsigned short* __restrict__ dst, int n) {
    for (int idx = blockIdx.x * blockDim.x + threadIdx.x; idx < n; idx += gridDim.x * blockDim.x) {
        int c = idx & 31;
        int ro = idx >> 5;
        dst[idx] = (c < CD) ? f2bf(src[(size_t)ro * CD + c]) : (unsigned short)0;
    }
}

// cond0 [B][16][T] -> [col][32] bf16 (upper 16 zero), col = b*T + t
__global__ void k_cvt_cond0(const float* __restrict__ c0, unsigned short* __restrict__ dst, int n) {
    for (int idx = blockIdx.x * blockDim.x + threadIdx.x; idx < n; idx += gridDim.x * blockDim.x) {
        int c   = idx & 31;
        int col = idx >> 5;
        int bb  = col >> 12;
        int tt  = col & (TLEN - 1);
        dst[idx] = (c < CD) ? f2bf(c0[((size_t)bb * CD + c) * TLEN + tt]) : (unsigned short)0;
    }
}

// ---------------- init causal conv (1 -> 512 channels, K=3, d=1) ----------------
__global__ __launch_bounds__(256)
void k_init_conv(const float* __restrict__ x, const float* __restrict__ w,
                 const float* __restrict__ b, float* __restrict__ feat,
                 unsigned short* __restrict__ featb) {
    int col = blockIdx.x;
    int bb = col >> 12, tt = col & (TLEN - 1);
    float x0 = (tt >= 2) ? x[bb * TLEN + tt - 2] : 0.f;
    float x1 = (tt >= 1) ? x[bb * TLEN + tt - 1] : 0.f;
    float x2 = x[bb * TLEN + tt];
#pragma unroll
    for (int j = 0; j < 2; ++j) {
        int o = threadIdx.x * 2 + j;
        float v = b[o] + w[o * 3 + 0] * x0 + w[o * 3 + 1] * x1 + w[o * 3 + 2] * x2;
        feat[(size_t)col * RW + o]  = v;
        featb[(size_t)col * RW + o] = f2bf(v);
    }
}

// ---------------- generic 256-row WMMA GEMM: C[col][256] = A[256][Kd] * B[col][Kd] (+bias) ----
// Bsrc is fp32 [col][Kd]; optional relu on B; optional transposed store into [B][256][T].
__global__ __launch_bounds__(128)
void k_gemm256(const unsigned short* __restrict__ A, const float* __restrict__ Bsrc,
               const float* __restrict__ bias, float* __restrict__ Cdst,
               int Kd, int reluB, int transT) {
    __shared__ unsigned short Bt[16 * LPAD];
    const int tid = threadIdx.x;
    const int colbase = blockIdx.x * 16;

    // stage bf16 B tile
    const int kch = Kd >> 3;
    for (int idx = tid; idx < 16 * kch; idx += 128) {
        int c  = idx / kch;
        int ch = (idx % kch) * 8;
        const float* sp = &Bsrc[((size_t)(colbase + c)) * Kd + ch];
        v8us o;
#pragma unroll
        for (int j = 0; j < 8; ++j) {
            float v = sp[j];
            if (reluB) v = v > 0.f ? v : 0.f;
            o[j] = f2bf(v);
        }
        *(v8us*)&Bt[c * LPAD + ch] = o;
    }
    __syncthreads();

    const int wid = tid >> 5, lane = tid & 31;
    const int n = lane & 15, kg = lane >> 4;

    v8f acc[4];
#pragma unroll
    for (int j = 0; j < 4; ++j) acc[j] = zero8();

    for (int ks = 0; ks < Kd; ks += 32) {
        const unsigned short* bp = &Bt[n * LPAD + ks + kg * 8];
        v16bf b = ldfrag(bp, bp + 16);
#pragma unroll
        for (int j = 0; j < 4; ++j) {
            int row = (wid * 4 + j) * 16 + n;
            const unsigned short* ap = &A[(size_t)row * Kd + ks + kg * 8];
            acc[j] = wmma_bf16(ldfrag(ap, ap + 16), b, acc[j]);
        }
    }

    const int col = colbase + n;
#pragma unroll
    for (int j = 0; j < 4; ++j) {
        int m0 = (wid * 4 + j) * 16 + kg * 8;
#pragma unroll
        for (int jj = 0; jj < 8; ++jj) {
            float v = acc[j][jj] + (bias ? bias[m0 + jj] : 0.f);
            if (!transT) {
                Cdst[(size_t)col * 256 + m0 + jj] = v;
            } else {
                int bb = col >> 12, tt = col & (TLEN - 1);
                Cdst[((size_t)bb * 256 + m0 + jj) * TLEN + tt] = v;
            }
        }
    }
}

// ---------------- fused WaveNet layer ----------------
// f = Wf*feat(shifted taps) + Wfc*cond ; g likewise ; r = sigmoid(f)*tanh(g)
// feat_out = feat_in + Wr*r ; skip += Ws*r
__global__ __launch_bounds__(256)
void k_layer(const unsigned short* __restrict__ wf, const unsigned short* __restrict__ wg,
             const unsigned short* __restrict__ wfc, const unsigned short* __restrict__ wgc,
             const unsigned short* __restrict__ wr, const unsigned short* __restrict__ wsk,
             const float* __restrict__ feat_in, const unsigned short* __restrict__ featb_in,
             float* __restrict__ feat_out, unsigned short* __restrict__ featb_out,
             float* __restrict__ skip, const unsigned short* __restrict__ cbp, int dil) {
    __shared__ unsigned short smem[3 * 16 * LPAD];   // 3 shifted B tiles; later aliased by r tile
    __shared__ unsigned short Bc[16 * 32];           // cond tile (K padded to 32)

    const int tid = threadIdx.x;
    const int colbase = blockIdx.x * 16;

    // stage the 3 dilation-shifted feature tiles (causal zero fill, 16B vectors)
    for (int idx = tid; idx < 3 * 16 * 64; idx += 256) {
        int s   = idx / (16 * 64);
        int rem = idx - s * 16 * 64;
        int c   = rem >> 6;
        int ch  = (rem & 63) * 8;
        int col = colbase + c;
        int shift = (2 - s) * dil;
        int tt = col & (TLEN - 1);
        v8us v = {0, 0, 0, 0, 0, 0, 0, 0};
        if (tt >= shift) v = *(const v8us*)&featb_in[(size_t)(col - shift) * RW + ch];
        *(v8us*)&smem[(s * 16 + c) * LPAD + ch] = v;
    }
    for (int idx = tid; idx < 64; idx += 256) {   // 16*32/8 chunks
        int c = idx >> 2;
        int ch = (idx & 3) * 8;
        *(v8us*)&Bc[c * 32 + ch] = *(const v8us*)&cbp[(size_t)(colbase + c) * 32 + ch];
    }
    __syncthreads();

    const int wid = tid >> 5, lane = tid & 31;
    const int n = lane & 15, kg = lane >> 4;

    // ---- pass 1: f and g ----
    v8f fa[4], ga[4];
#pragma unroll
    for (int j = 0; j < 4; ++j) { fa[j] = zero8(); ga[j] = zero8(); }

#pragma unroll
    for (int s = 0; s < 3; ++s) {
        const unsigned short* brow = &smem[(s * 16 + n) * LPAD + kg * 8];
        const unsigned short* wfs = wf + (size_t)s * RW * RW;
        const unsigned short* wgs = wg + (size_t)s * RW * RW;
        for (int ks = 0; ks < RW; ks += 32) {
            v16bf b = ldfrag(brow + ks, brow + ks + 16);
#pragma unroll
            for (int j = 0; j < 4; ++j) {
                int row = (wid * 4 + j) * 16 + n;
                const unsigned short* af = &wfs[(size_t)row * RW + ks + kg * 8];
                fa[j] = wmma_bf16(ldfrag(af, af + 16), b, fa[j]);
                const unsigned short* ag = &wgs[(size_t)row * RW + ks + kg * 8];
                ga[j] = wmma_bf16(ldfrag(ag, ag + 16), b, ga[j]);
            }
        }
    }
    {   // conditional contribution: one zero-padded K=32 step
        const unsigned short* bp = &Bc[n * 32 + kg * 8];
        v16bf b = ldfrag(bp, bp + 16);
#pragma unroll
        for (int j = 0; j < 4; ++j) {
            int row = (wid * 4 + j) * 16 + n;
            const unsigned short* af = &wfc[(size_t)row * 32 + kg * 8];
            fa[j] = wmma_bf16(ldfrag(af, af + 16), b, fa[j]);
            const unsigned short* ag = &wgc[(size_t)row * 32 + kg * 8];
            ga[j] = wmma_bf16(ldfrag(ag, ag + 16), b, ga[j]);
        }
    }

    __syncthreads();   // all waves done reading staged B tiles

    // ---- gating, write r tile (channel-contiguous bf16) into smem ----
#pragma unroll
    for (int j = 0; j < 4; ++j) {
        v8us r;
#pragma unroll
        for (int jj = 0; jj < 8; ++jj) {
            float f = fa[j][jj], g = ga[j][jj];
            float sg = 1.0f / (1.0f + __expf(-f));
            float e2 = __expf(2.0f * g);
            float th = 1.0f - 2.0f / (e2 + 1.0f);
            r[jj] = f2bf(sg * th);
        }
        *(v8us*)&smem[n * LPAD + (wid * 4 + j) * 16 + kg * 8] = r;
    }
    __syncthreads();

    // ---- pass 2: residual (4 tiles) + skip (2 tiles) per wave ----
    v8f acc[6];
#pragma unroll
    for (int j = 0; j < 6; ++j) acc[j] = zero8();

    for (int ks = 0; ks < RW; ks += 32) {
        const unsigned short* bp = &smem[n * LPAD + ks + kg * 8];
        v16bf b = ldfrag(bp, bp + 16);
#pragma unroll
        for (int j = 0; j < 4; ++j) {
            int row = (wid * 4 + j) * 16 + n;
            const unsigned short* ap = &wr[(size_t)row * RW + ks + kg * 8];
            acc[j] = wmma_bf16(ldfrag(ap, ap + 16), b, acc[j]);
        }
#pragma unroll
        for (int j = 0; j < 2; ++j) {
            int row = (wid * 2 + j) * 16 + n;
            const unsigned short* ap = &wsk[(size_t)row * RW + ks + kg * 8];
            acc[4 + j] = wmma_bf16(ldfrag(ap, ap + 16), b, acc[4 + j]);
        }
    }

    const int col = colbase + n;
#pragma unroll
    for (int j = 0; j < 4; ++j) {           // residual: feat_out = feat_in + acc
        int m0 = (wid * 4 + j) * 16 + kg * 8;
        const float* fi = &feat_in[(size_t)col * RW + m0];
        float* fo = &feat_out[(size_t)col * RW + m0];
        v8us fb;
#pragma unroll
        for (int jj = 0; jj < 8; ++jj) {
            float v = fi[jj] + acc[j][jj];
            fo[jj] = v;
            fb[jj] = f2bf(v);
        }
        *(v8us*)&featb_out[(size_t)col * RW + m0] = fb;
    }
#pragma unroll
    for (int j = 0; j < 2; ++j) {           // skip accumulate (columns WG-exclusive)
        int m0 = (wid * 2 + j) * 16 + kg * 8;
        float* sp = &skip[(size_t)col * SW + m0];
#pragma unroll
        for (int jj = 0; jj < 8; ++jj) sp[jj] += acc[4 + j][jj];
    }
}

// ---------------- host launch ----------------
extern "C" void kernel_launch(void* const* d_in, const int* in_sizes, int n_in,
                              void* d_out, int out_size, void* d_ws, size_t ws_size,
                              hipStream_t stream) {
    (void)in_sizes; (void)n_in; (void)out_size; (void)ws_size;
    const float* x           = (const float*)d_in[0];
    const float* cond0       = (const float*)d_in[1];
    const float* init_w      = (const float*)d_in[2];
    const float* init_b      = (const float*)d_in[3];
    const float* init_skip_w = (const float*)d_in[4];
    const float* init_skip_b = (const float*)d_in[5];
    const float* filter_w    = (const float*)d_in[6];
    const float* gate_w      = (const float*)d_in[7];
    const float* skip_w      = (const float*)d_in[8];
    const float* feat_w      = (const float*)d_in[9];
    const float* fcond_w     = (const float*)d_in[10];
    const float* gcond_w     = (const float*)d_in[11];
    const float* fs_w        = (const float*)d_in[12];
    const float* fs_b        = (const float*)d_in[13];
    const float* fin_w       = (const float*)d_in[14];
    const float* fin_b       = (const float*)d_in[15];

    size_t off = 0;
    auto carve = [&](size_t bytes) -> void* {
        void* p = (char*)d_ws + off;
        off += (bytes + 255) & ~(size_t)255;
        return p;
    };
    float*          featA  = (float*)carve((size_t)RW * NC * 4);
    float*          featB  = (float*)carve((size_t)RW * NC * 4);
    unsigned short* featbA = (unsigned short*)carve((size_t)RW * NC * 2);
    unsigned short* featbB = (unsigned short*)carve((size_t)RW * NC * 2);
    float*          skip32 = (float*)carve((size_t)SW * NC * 4);
    float*          skip2  = (float*)carve((size_t)SW * NC * 4);
    unsigned short* cbp    = (unsigned short*)carve((size_t)NC * 32 * 2);
    unsigned short* wfb    = (unsigned short*)carve((size_t)NLAY * 3 * RW * RW * 2);
    unsigned short* wgb    = (unsigned short*)carve((size_t)NLAY * 3 * RW * RW * 2);
    unsigned short* wfcb   = (unsigned short*)carve((size_t)NLAY * RW * 32 * 2);
    unsigned short* wgcb   = (unsigned short*)carve((size_t)NLAY * RW * 32 * 2);
    unsigned short* wrb    = (unsigned short*)carve((size_t)NLAY * RW * RW * 2);
    unsigned short* wskb   = (unsigned short*)carve((size_t)NLAY * SW * RW * 2);
    unsigned short* wisb   = (unsigned short*)carve((size_t)SW * RW * 2);
    unsigned short* wfsb   = (unsigned short*)carve((size_t)SW * SW * 2);
    unsigned short* wfinb  = (unsigned short*)carve((size_t)OC * SW * 2);

    const int CVB = 2048, CVT = 256;
    // weight / input conversions (every call; no cached state)
    k_cvt_conv <<<CVB, CVT, 0, stream>>>(filter_w, wfb, NLAY * 3 * RW * RW);
    k_cvt_conv <<<CVB, CVT, 0, stream>>>(gate_w,   wgb, NLAY * 3 * RW * RW);
    k_cvt_cond <<<CVB, CVT, 0, stream>>>(fcond_w, wfcb, NLAY * RW * 32);
    k_cvt_cond <<<CVB, CVT, 0, stream>>>(gcond_w, wgcb, NLAY * RW * 32);
    k_cvt_flat <<<CVB, CVT, 0, stream>>>(feat_w,      wrb,   NLAY * RW * RW);
    k_cvt_flat <<<CVB, CVT, 0, stream>>>(skip_w,      wskb,  NLAY * SW * RW);
    k_cvt_flat <<<CVB, CVT, 0, stream>>>(init_skip_w, wisb,  SW * RW);
    k_cvt_flat <<<CVB, CVT, 0, stream>>>(fs_w,        wfsb,  SW * SW);
    k_cvt_flat <<<CVB, CVT, 0, stream>>>(fin_w,       wfinb, OC * SW);
    k_cvt_cond0<<<CVB, CVT, 0, stream>>>(cond0, cbp, NC * 32);

    // init conv + init skip projection
    k_init_conv<<<NC, 256, 0, stream>>>(x, init_w, init_b, featA, featbA);
    k_gemm256  <<<NC / 16, 128, 0, stream>>>(wisb, featA, init_skip_b, skip32,
                                             RW, /*relu=*/0, /*trans=*/0);

    // 30 fused layers, ping-pong feature buffers
    for (int l = 0; l < NLAY; ++l) {
        int dl  = 2 << (l % 10);   // dilations 2,4,...,1024 per block
        int ping = l & 1;
        float*          fi  = ping ? featB  : featA;
        unsigned short* fbi = ping ? featbB : featbA;
        float*          fo  = ping ? featA  : featB;
        unsigned short* fbo = ping ? featbA : featbB;
        k_layer<<<NC / 16, 256, 0, stream>>>(
            wfb  + (size_t)l * 3 * RW * RW,
            wgb  + (size_t)l * 3 * RW * RW,
            wfcb + (size_t)l * RW * 32,
            wgcb + (size_t)l * RW * 32,
            wrb  + (size_t)l * RW * RW,
            wskb + (size_t)l * SW * RW,
            fi, fbi, fo, fbo, skip32, cbp, dl);
    }

    // final: skip2 = fs_w @ relu(skip) + fs_b ; out = fin_w @ relu(skip2) + fin_b
    k_gemm256<<<NC / 16, 128, 0, stream>>>(wfsb,  skip32, fs_b,  skip2,
                                           SW, /*relu=*/1, /*trans=*/0);
    k_gemm256<<<NC / 16, 128, 0, stream>>>(wfinb, skip2,  fin_b, (float*)d_out,
                                           SW, /*relu=*/1, /*trans=*/1);
}
